// MyLLMDecoderLayer_91190745629053
// MI455X (gfx1250) — matrix operand order repaired and checked by reading, not compiled, not run on hardware
//
#include <hip/hip_runtime.h>
#include <hip/hip_bf16.h>
#include <math.h>

#define S_  2048
#define H_  2048
#define NH_ 16
#define HD_ 128
#define NKV_ 4
#define FF_ 8192

typedef __attribute__((ext_vector_type(16))) __bf16 v16bf;
typedef __attribute__((ext_vector_type(8)))  float  v8f;

struct alignas(32) Frag32B { uint4 lo, hi; };

static __device__ __forceinline__ __bf16 f2bf(float x) { return (__bf16)x; }

// ---------------------------------------------------------------------------
// WMMA GEMM: C[M,N](f32) = A[M,K](bf16 row-major, lda) x Bt[N,K](bf16 row-major, ldb)^T
// Block tile 128x128, BK=32, 256 threads = 8 waves in 2(M) x 4(N); each wave
// computes a 64x32 region = 4x2 tiles of v_wmma_f32_16x16x32_bf16.
// Double-buffered LDS (one barrier per k-step); global prefetch 2 steps ahead.
// MODE 0: C = acc + bias[col]        MODE 1: C = causal/mask ? exp(acc*scale) : 0
// ---------------------------------------------------------------------------
template<int MODE>
__global__ __launch_bounds__(256) void wmma_gemm(
    const __bf16* __restrict__ A, int lda,
    const __bf16* __restrict__ Bt, int ldb,
    float* __restrict__ C, int ldc,
    const float* __restrict__ bias,
    const int* __restrict__ mask,
    float scale, int K)
{
  // 128 rows x 32 cols (+8 pad) per buffer; row stride 80B (16B aligned)
  __shared__ __bf16 sA[2][128 * 40];
  __shared__ __bf16 sB[2][128 * 40];

  const int tid  = threadIdx.x;
  const int lane = tid & 31;
  const int wave = tid >> 5;
  const int wm   = wave >> 2;            // 0..1
  const int wn   = wave & 3;             // 0..3
  const int row0 = blockIdx.y * 128;
  const int col0 = blockIdx.x * 128;

  const int ldRow = tid >> 1;            // 0..127
  const int ldCol = (tid & 1) * 16;      // 0 or 16 (elements)
  const int m16   = lane & 15;
  const int hh    = lane >> 4;           // 0 or 1

  const __bf16* gArow = A  + (size_t)(row0 + ldRow) * lda + ldCol;
  const __bf16* gBrow = Bt + (size_t)(col0 + ldRow) * ldb + ldCol;

  const v8f vz = {0.f,0.f,0.f,0.f,0.f,0.f,0.f,0.f};
  v8f acc[4][2];
  #pragma unroll
  for (int i = 0; i < 4; ++i)
    #pragma unroll
    for (int j = 0; j < 2; ++j) acc[i][j] = vz;

  // preload k-step 0 into buffer 0 (each thread moves 16 bf16 = two uint4)
  {
    const uint4* gA = (const uint4*)gArow;
    const uint4* gB = (const uint4*)gBrow;
    uint4 a0 = gA[0], a1 = gA[1], b0 = gB[0], b1 = gB[1];
    uint4* dA = (uint4*)&sA[0][ldRow * 40 + ldCol];
    uint4* dB = (uint4*)&sB[0][ldRow * 40 + ldCol];
    dA[0] = a0; dA[1] = a1;
    dB[0] = b0; dB[1] = b1;
  }
  __syncthreads();

  int cur = 0;
  for (int k0 = 0; k0 < K; k0 += 32) {
    const int knext = k0 + 32;

    // prefetch 2 k-steps ahead (lowers to global_prefetch_b8)
    if (k0 + 64 < K) {
      __builtin_prefetch(gArow + k0 + 64, 0, 1);
      __builtin_prefetch(gBrow + k0 + 64, 0, 1);
    }

    // issue next tile's global loads early (into registers)
    uint4 a0 = {}, a1 = {}, b0 = {}, b1 = {};
    if (knext < K) {
      const uint4* gA = (const uint4*)(gArow + knext);
      const uint4* gB = (const uint4*)(gBrow + knext);
      a0 = gA[0]; a1 = gA[1]; b0 = gB[0]; b1 = gB[1];
    }

    // fragments from current buffer
    v16bf afr[4], bfr[2];
    #pragma unroll
    for (int i = 0; i < 4; ++i) {
      // A 16x32 bf16 layout: elems 0..7 -> K=h*8.., elems 8..15 -> K=16+h*8..
      const __bf16* p = &sA[cur][(wm * 64 + i * 16 + m16) * 40];
      Frag32B f;
      f.lo = *(const uint4*)(p + hh * 8);
      f.hi = *(const uint4*)(p + 16 + hh * 8);
      afr[i] = __builtin_bit_cast(v16bf, f);
    }
    #pragma unroll
    for (int j = 0; j < 2; ++j) {
      // B 32x16 bf16 layout: elems 0..7 -> K=h*8.., elems 8..15 -> K=16+h*8..
      const __bf16* p = &sB[cur][(wn * 32 + j * 16 + m16) * 40 + hh * 8];
      Frag32B f;
      f.lo = *(const uint4*)(p);
      f.hi = *(const uint4*)(p + 16);
      bfr[j] = __builtin_bit_cast(v16bf, f);
    }
    #pragma unroll
    for (int i = 0; i < 4; ++i)
      #pragma unroll
      for (int j = 0; j < 2; ++j)
        acc[i][j] = __builtin_amdgcn_wmma_f32_16x16x32_bf16(
            false, afr[i], false, bfr[j], (short)0, acc[i][j], false, false);

    // stage next tile into the other buffer; single barrier per k-step
    if (knext < K) {
      uint4* dA = (uint4*)&sA[cur ^ 1][ldRow * 40 + ldCol];
      uint4* dB = (uint4*)&sB[cur ^ 1][ldRow * 40 + ldCol];
      dA[0] = a0; dA[1] = a1;
      dB[0] = b0; dB[1] = b1;
      __syncthreads();
      cur ^= 1;
    }
  }

  // Epilogue. D layout: lane -> col = m16; VGPR e -> row = h*8 + e.
  #pragma unroll
  for (int j = 0; j < 2; ++j) {
    const int col = col0 + wn * 32 + j * 16 + m16;
    float bv = 0.f;
    bool  mok = true;
    if (MODE == 0) { if (bias) bv = bias[col]; }
    else           { mok = (mask[col] != 0); }
    #pragma unroll
    for (int i = 0; i < 4; ++i) {
      const int rbase = row0 + wm * 64 + i * 16 + hh * 8;
      #pragma unroll
      for (int e = 0; e < 8; ++e) {
        const int r = rbase + e;
        float v = acc[i][j][e];
        if (MODE == 0) v += bv;
        else           v = (mok && col <= r) ? expf(v * scale) : 0.f;
        C[(size_t)r * ldc + col] = v;
      }
    }
  }
}

// ---------------------------------------------------------------------------
// Transpose + fp32->bf16 convert: out[n*ldout + r] = (bf16) in[r*ldin + n]
// grid: (Ncols/32, Rrows/32, batch), block (32,8)
// ---------------------------------------------------------------------------
__global__ __launch_bounds__(256) void transpose_f32_bf16(
    const float* __restrict__ in, int ldin, long long inBatch,
    __bf16* __restrict__ out, int ldout, long long outBatch)
{
  __shared__ float t[32][33];
  const int tx = threadIdx.x, ty = threadIdx.y;
  const int nb = blockIdx.x * 32, rb = blockIdx.y * 32;
  in  += (size_t)blockIdx.z * inBatch;
  out += (size_t)blockIdx.z * outBatch;
  #pragma unroll
  for (int j = ty; j < 32; j += 8)
    t[j][tx] = in[(size_t)(rb + j) * ldin + nb + tx];
  __syncthreads();
  #pragma unroll
  for (int j = ty; j < 32; j += 8)
    out[(size_t)(nb + j) * ldout + rb + tx] = f2bf(t[tx][j]);
}

// RMSNorm: one block per row, fp32 in -> bf16 out
__global__ __launch_bounds__(256) void rmsnorm_bf16(
    const float* __restrict__ x, const float* __restrict__ gamma,
    __bf16* __restrict__ out, int n)
{
  __shared__ float red[256];
  const int row = blockIdx.x, tid = threadIdx.x;
  const float* xr = x + (size_t)row * n;
  float s = 0.f;
  for (int c = tid; c < n; c += 256) { float v = xr[c]; s += v * v; }
  red[tid] = s; __syncthreads();
  for (int off = 128; off > 0; off >>= 1) {
    if (tid < off) red[tid] += red[tid + off];
    __syncthreads();
  }
  const float inv = 1.f / sqrtf(red[0] / n + 1e-5f);
  for (int c = tid; c < n; c += 256)
    out[(size_t)row * n + c] = f2bf(xr[c] * inv * gamma[c]);
}

// Faithful non-standard softmax row fixup:
// p already = exp(score) with causal+padding baked to 0.
// w = (p - max) / (sum_{mask}(p - max) + 1e-6); masked cols -> 0.   bf16 out.
__global__ __launch_bounds__(256) void attn_rownorm(
    const float* __restrict__ P, const int* __restrict__ mask,
    __bf16* __restrict__ Pn, int n)
{
  __shared__ float rmx[256], rsm[256], rct[256];
  const int row = blockIdx.x, tid = threadIdx.x;
  const float* pr = P + (size_t)row * n;
  float mx = -1e30f, sm = 0.f, ct = 0.f;
  for (int c = tid; c < n; c += 256) {
    float p = pr[c];
    mx = fmaxf(mx, p);
    if (mask[c] != 0) { sm += p; ct += 1.f; }
  }
  rmx[tid] = mx; rsm[tid] = sm; rct[tid] = ct; __syncthreads();
  for (int off = 128; off > 0; off >>= 1) {
    if (tid < off) {
      rmx[tid] = fmaxf(rmx[tid], rmx[tid + off]);
      rsm[tid] += rsm[tid + off];
      rct[tid] += rct[tid + off];
    }
    __syncthreads();
  }
  mx = rmx[0];
  const float inv = 1.f / ((rsm[0] - rct[0] * mx) + 1e-6f);
  for (int c = tid; c < n; c += 256)
    Pn[(size_t)row * n + c] = (mask[c] != 0) ? f2bf((pr[c] - mx) * inv)
                                             : f2bf(0.f);
}

// elementwise kernels (n is always a multiple of 256; 1 elem/thread)
__global__ void k_f32_to_bf16(const float* __restrict__ in, __bf16* __restrict__ out) {
  size_t i = (size_t)blockIdx.x * 256 + threadIdx.x;
  out[i] = f2bf(in[i]);
}
__global__ void k_posmul_bf16(const float* __restrict__ in, const float* __restrict__ pos,
                              __bf16* __restrict__ out, int cols) {
  size_t i = (size_t)blockIdx.x * 256 + threadIdx.x;
  int s = (int)(i / cols);
  int d = ((int)(i % cols)) & (HD_ - 1);
  out[i] = f2bf(in[i] * pos[s * HD_ + d]);
}
__global__ void k_add_inplace(float* __restrict__ a, const float* __restrict__ b) {
  size_t i = (size_t)blockIdx.x * 256 + threadIdx.x;
  a[i] += b[i];
}
__global__ void k_swiglu_bf16(const float* __restrict__ x, const float* __restrict__ g,
                              __bf16* __restrict__ out) {
  size_t i = (size_t)blockIdx.x * 256 + threadIdx.x;
  float u = x[i];
  out[i] = f2bf(u * (1.f / (1.f + expf(-u))) * g[i]);
}
__global__ void k_swiglu_add_out(const float* __restrict__ x, const float* __restrict__ g,
                                 const float* __restrict__ res, float* __restrict__ out) {
  size_t i = (size_t)blockIdx.x * 256 + threadIdx.x;
  float u = x[i];
  out[i] = u * (1.f / (1.f + expf(-u))) * g[i] + res[i];
}

// ---------------------------------------------------------------------------
extern "C" void kernel_launch(void* const* d_in, const int* in_sizes, int n_in,
                              void* d_out, int out_size, void* d_ws, size_t ws_size,
                              hipStream_t stream) {
  (void)in_sizes; (void)n_in; (void)out_size; (void)ws_size;

  const float* hs    = (const float*)d_in[0];
  const int*   amask = (const int*)  d_in[1];
  const float* pos   = (const float*)d_in[2];
  const float* Wq  = (const float*)d_in[3];  const float* bq  = (const float*)d_in[4];
  const float* Wk  = (const float*)d_in[5];  const float* bk  = (const float*)d_in[6];
  const float* Wv  = (const float*)d_in[7];  const float* bv  = (const float*)d_in[8];
  const float* Wo  = (const float*)d_in[9];  const float* bo  = (const float*)d_in[10];
  const float* gin = (const float*)d_in[11]; const float* gout= (const float*)d_in[12];
  const float* Wup = (const float*)d_in[13]; const float* bup = (const float*)d_in[14];
  const float* Wdn = (const float*)d_in[15]; const float* bdn = (const float*)d_in[16];
  const float* Wsu = (const float*)d_in[17]; const float* bsu = (const float*)d_in[18];
  const float* Wsd = (const float*)d_in[19]; const float* bsd = (const float*)d_in[20];
  float* outp = (float*)d_out;

  char* base = (char*)d_ws;
  size_t off = 0;
  auto alloc = [&](size_t bytes) -> void* {
    off = (off + 255) & ~(size_t)255;
    void* p = base + off;
    off += bytes;
    return p;
  };

  // bf16 transposed weights [N][K]
  __bf16* WqT = (__bf16*)alloc((size_t)H_ * H_ * 2);
  __bf16* WkT = (__bf16*)alloc((size_t)(NKV_*HD_) * H_ * 2);
  __bf16* WvT = (__bf16*)alloc((size_t)(NKV_*HD_) * H_ * 2);
  __bf16* WoT = (__bf16*)alloc((size_t)H_ * H_ * 2);
  __bf16* WupT= (__bf16*)alloc((size_t)FF_ * H_ * 2);
  __bf16* WdnT= (__bf16*)alloc((size_t)H_ * FF_ * 2);
  __bf16* WsuT= (__bf16*)alloc((size_t)FF_ * FF_ * 2);
  __bf16* WsdT= (__bf16*)alloc((size_t)H_ * H_ * 2);
  // activations
  __bf16* xn1 = (__bf16*)alloc((size_t)S_ * H_ * 2);
  float*  qf  = (float*) alloc((size_t)S_ * H_ * 4);
  float*  kf  = (float*) alloc((size_t)S_ * (NKV_*HD_) * 4);
  float*  vf  = (float*) alloc((size_t)S_ * (NKV_*HD_) * 4);
  __bf16* qb  = (__bf16*)alloc((size_t)S_ * H_ * 2);
  __bf16* kbf = (__bf16*)alloc((size_t)S_ * (NKV_*HD_) * 2);
  __bf16* vt  = (__bf16*)alloc((size_t)NKV_ * HD_ * S_ * 2);
  float*  P   = (float*) alloc((size_t)S_ * S_ * 4);
  __bf16* Pn  = (__bf16*)alloc((size_t)S_ * S_ * 2);
  float*  O   = (float*) alloc((size_t)S_ * H_ * 4);
  __bf16* Ob  = (__bf16*)alloc((size_t)S_ * H_ * 2);
  float*  x2  = (float*) alloc((size_t)S_ * H_ * 4);   // attn out, then +residual
  __bf16* xn2 = (__bf16*)alloc((size_t)S_ * H_ * 2);
  float*  up  = (float*) alloc((size_t)S_ * FF_ * 4);
  __bf16* upb = (__bf16*)alloc((size_t)S_ * FF_ * 2);
  float*  g1  = (float*) alloc((size_t)S_ * FF_ * 4);
  __bf16* s1  = (__bf16*)alloc((size_t)S_ * FF_ * 2);
  float*  dn  = (float*) alloc((size_t)S_ * H_ * 4);
  __bf16* dnb = (__bf16*)alloc((size_t)S_ * H_ * 2);
  float*  g2  = (float*) alloc((size_t)S_ * H_ * 4);

  const dim3 T8(32, 8);
  // weight transpose-convert (prep per call)
  transpose_f32_bf16<<<dim3(H_/32,  H_/32,  1), T8, 0, stream>>>(Wq,  H_,        0, WqT,  H_,  0);
  transpose_f32_bf16<<<dim3((NKV_*HD_)/32, H_/32, 1), T8, 0, stream>>>(Wk, NKV_*HD_, 0, WkT, H_, 0);
  transpose_f32_bf16<<<dim3((NKV_*HD_)/32, H_/32, 1), T8, 0, stream>>>(Wv, NKV_*HD_, 0, WvT, H_, 0);
  transpose_f32_bf16<<<dim3(H_/32,  H_/32,  1), T8, 0, stream>>>(Wo,  H_,        0, WoT,  H_,  0);
  transpose_f32_bf16<<<dim3(FF_/32, H_/32,  1), T8, 0, stream>>>(Wup, FF_,       0, WupT, H_,  0);
  transpose_f32_bf16<<<dim3(H_/32,  FF_/32, 1), T8, 0, stream>>>(Wdn, H_,        0, WdnT, FF_, 0);
  transpose_f32_bf16<<<dim3(FF_/32, FF_/32, 1), T8, 0, stream>>>(Wsu, FF_,       0, WsuT, FF_, 0);
  transpose_f32_bf16<<<dim3(H_/32,  H_/32,  1), T8, 0, stream>>>(Wsd, H_,        0, WsdT, H_,  0);

  // ---- attention branch ----
  rmsnorm_bf16<<<S_, 256, 0, stream>>>(hs, gin, xn1, H_);

  wmma_gemm<0><<<dim3(H_/128, S_/128), 256, 0, stream>>>(xn1, H_, WqT, H_, qf, H_, bq, nullptr, 1.f, H_);
  wmma_gemm<0><<<dim3((NKV_*HD_)/128, S_/128), 256, 0, stream>>>(xn1, H_, WkT, H_, kf, NKV_*HD_, bk, nullptr, 1.f, H_);
  wmma_gemm<0><<<dim3((NKV_*HD_)/128, S_/128), 256, 0, stream>>>(xn1, H_, WvT, H_, vf, NKV_*HD_, bv, nullptr, 1.f, H_);

  k_posmul_bf16<<<(S_*H_)/256, 256, 0, stream>>>(qf, pos, qb, H_);
  k_posmul_bf16<<<(S_*(NKV_*HD_))/256, 256, 0, stream>>>(kf, pos, kbf, NKV_*HD_);
  // vt[h][d][s] = v[s][h*HD+d]
  transpose_f32_bf16<<<dim3(HD_/32, S_/32, NKV_), T8, 0, stream>>>(
      vf, NKV_*HD_, (long long)HD_, vt, S_, (long long)HD_ * S_);

  const float sc = 1.f / (float)HD_;
  for (int h = 0; h < NH_; ++h) {
    const int kv = h / (NH_ / NKV_);
    // scores: exp((q_h @ k_h^T)/HD) with causal+padding mask baked in
    wmma_gemm<1><<<dim3(S_/128, S_/128), 256, 0, stream>>>(
        qb + h * HD_, H_, kbf + kv * HD_, NKV_*HD_, P, S_, nullptr, amask, sc, HD_);
    attn_rownorm<<<S_, 256, 0, stream>>>(P, amask, Pn, S_);
    // O[:, h*HD:(h+1)*HD] = Pn @ V_kv
    wmma_gemm<0><<<dim3(HD_/128, S_/128), 256, 0, stream>>>(
        Pn, S_, vt + (size_t)kv * HD_ * S_, S_, O + h * HD_, H_, nullptr, nullptr, 1.f, S_);
  }

  k_f32_to_bf16<<<(S_*H_)/256, 256, 0, stream>>>(O, Ob);
  wmma_gemm<0><<<dim3(H_/128, S_/128), 256, 0, stream>>>(Ob, H_, WoT, H_, x2, H_, bo, nullptr, 1.f, H_);
  k_add_inplace<<<(S_*H_)/256, 256, 0, stream>>>(x2, hs);   // x2 = attn + residual

  // ---- FFN branch ----
  rmsnorm_bf16<<<S_, 256, 0, stream>>>(x2, gout, xn2, H_);
  wmma_gemm<0><<<dim3(FF_/128, S_/128), 256, 0, stream>>>(xn2, H_, WupT, H_, up, FF_, bup, nullptr, 1.f, H_);
  k_f32_to_bf16<<<(S_*FF_)/256, 256, 0, stream>>>(up, upb);
  wmma_gemm<0><<<dim3(FF_/128, S_/128), 256, 0, stream>>>(upb, FF_, WsuT, FF_, g1, FF_, bsu, nullptr, 1.f, FF_);
  k_swiglu_bf16<<<(S_*FF_)/256, 256, 0, stream>>>(up, g1, s1);
  wmma_gemm<0><<<dim3(H_/128, S_/128), 256, 0, stream>>>(s1, FF_, WdnT, FF_, dn, H_, bdn, nullptr, 1.f, FF_);
  k_f32_to_bf16<<<(S_*H_)/256, 256, 0, stream>>>(dn, dnb);
  wmma_gemm<0><<<dim3(H_/128, S_/128), 256, 0, stream>>>(dnb, H_, WsdT, H_, g2, H_, bsd, nullptr, 1.f, H_);
  k_swiglu_add_out<<<(S_*H_)/256, 256, 0, stream>>>(dn, g2, x2, outp);
}
